// GraphNetworkMetaLayer_25598005084726
// MI455X (gfx1250) — matrix operand order repaired and checked by reading, not compiled, not run on hardware
//
#include <hip/hip_runtime.h>

#define NN 100000
#define EE 1600000
#define FF 32

// padded LDS row strides (bf16 elements) to avoid bank conflicts
#define W1TS 136   // edge W1^T rows of 128 k, padded
#define W2TS 136   // W2^T rows of 128 k, padded
#define HS   136   // hidden scratch rows of 128, padded
#define W1NS 104   // node W1^T rows of 96 k, padded

typedef __attribute__((ext_vector_type(16))) __bf16 v16bf;
typedef __attribute__((ext_vector_type(8)))  float  v8f;

union AF { v16bf v; uint4 q[2]; };

__device__ __forceinline__ v8f wmma_bf16(v16bf a, v16bf b, v8f c) {
  return __builtin_amdgcn_wmma_f32_16x16x32_bf16(false, a, false, b, (short)0, c,
                                                 false, false);
}

__device__ __forceinline__ void cvt16(AF& dst, const float* f, float scale) {
#pragma unroll
  for (int j = 0; j < 16; ++j) dst.v[j] = (__bf16)(f[j] * scale);
}

__device__ __forceinline__ void load16(float* f, const float* p, int hi) {
  // A-layout for 16-bit 16x32 fragment: lane(hi=0): K 0..7 & 16..23,
  // lane(hi=1): K 8..15 & 24..31  -> two contiguous 32B chunks
  float4 f0 = *(const float4*)(p + 8 * hi);
  float4 f1 = *(const float4*)(p + 8 * hi + 4);
  float4 f2 = *(const float4*)(p + 16 + 8 * hi);
  float4 f3 = *(const float4*)(p + 16 + 8 * hi + 4);
  f[0] = f0.x; f[1] = f0.y; f[2]  = f0.z; f[3]  = f0.w;
  f[4] = f1.x; f[5] = f1.y; f[6]  = f1.z; f[7]  = f1.w;
  f[8] = f2.x; f[9] = f2.y; f[10] = f2.z; f[11] = f2.w;
  f[12] = f3.x; f[13] = f3.y; f[14] = f3.z; f[15] = f3.w;
}

// ---------------------------------------------------------------- zero scratch
__global__ void zero_ws(float* __restrict__ p, long n) {
  long i = (long)blockIdx.x * blockDim.x + threadIdx.x;
  long stride = (long)gridDim.x * blockDim.x;
  for (; i < n; i += stride) p[i] = 0.0f;
}

// ---------------------------------------------------------------- edge model
// in = [x[src], x[dst], edge_attr, global] (128) -> 128 relu -> 32
__global__ __launch_bounds__(256) void edge_mlp_kernel(
    const float* __restrict__ x, const int* __restrict__ ei,
    const float* __restrict__ ea, const float* __restrict__ ga,
    const float* __restrict__ W1, const float* __restrict__ b1,
    const float* __restrict__ W2, const float* __restrict__ b2,
    float* __restrict__ eout, float* __restrict__ agg, float* __restrict__ cnt) {
  __shared__ alignas(16) __bf16 w1t[128 * W1TS];
  __shared__ alignas(16) __bf16 w2t[32 * W2TS];
  __shared__ alignas(16) __bf16 hid[8][16 * HS];
  __shared__ float sb1[128];
  __shared__ float sb2[32];
  __shared__ int   sdst[8][16];

  const int t = threadIdx.x;
  for (int idx = t; idx < 128 * 128; idx += 256) {   // W1[i][j] -> w1t[j][i]
    int i = idx >> 7, j = idx & 127;
    w1t[j * W1TS + i] = (__bf16)W1[idx];
  }
  for (int idx = t; idx < 128 * 32; idx += 256) {    // W2[k][n] -> w2t[n][k]
    int k = idx >> 5, n = idx & 31;
    w2t[n * W2TS + k] = (__bf16)W2[idx];
  }
  if (t < 128) sb1[t] = b1[t];
  if (t < 32)  sb2[t] = b2[t];
  __syncthreads();

  const int wave = t >> 5, lane = t & 31;
  const int m = lane & 15, hi = lane >> 4;
  const long e0 = ((long)blockIdx.x * 8 + wave) * 16;
  const int e = (int)e0 + m;
  const int src = ei[e];
  const int dst = ei[EE + e];
  if (hi == 0) sdst[wave][m] = dst;

  // ---- build A fragments (4 K-tiles of 32) in bf16
  AF a[4];
  {
    const float* bases[4] = {x + (long)src * FF, x + (long)dst * FF,
                             ea + (long)e * FF, ga};
#pragma unroll
    for (int kt = 0; kt < 4; ++kt) {
      float f[16];
      load16(f, bases[kt], hi);
      cvt16(a[kt], f, 1.0f);
    }
  }

  // ---- layer 1: 8 n-tiles x 4 k-tiles, relu, to LDS hidden (bf16)
  __bf16* hw = &hid[wave][0];
#pragma unroll
  for (int nt = 0; nt < 8; ++nt) {
    v8f c = {0.f, 0.f, 0.f, 0.f, 0.f, 0.f, 0.f, 0.f};
#pragma unroll
    for (int kt = 0; kt < 4; ++kt) {
      AF bfr;
      const uint4* bp = (const uint4*)&w1t[(16 * nt + m) * W1TS + 32 * kt + 8 * hi];
      bfr.q[0] = bp[0];
      bfr.q[1] = bp[2];
      c = wmma_bf16(a[kt].v, bfr.v, c);
    }
    const float bias = sb1[16 * nt + m];
#pragma unroll
    for (int r = 0; r < 8; ++r) {
      float h = c[r] + bias;
      h = h > 0.f ? h : 0.f;
      hw[(r + 8 * hi) * HS + 16 * nt + m] = (__bf16)h;
    }
  }
  asm volatile("" ::: "memory");  // keep LDS hidden writes before reads

  // ---- layer 2: 2 n-tiles x 4 k-tiles, bias, store + scatter atomics
#pragma unroll
  for (int nt = 0; nt < 2; ++nt) {
    v8f c = {0.f, 0.f, 0.f, 0.f, 0.f, 0.f, 0.f, 0.f};
#pragma unroll
    for (int kt = 0; kt < 4; ++kt) {
      AF av, bv;
      av.q[0] = *(const uint4*)&hw[m * HS + 32 * kt + 8 * hi];
      av.q[1] = *(const uint4*)&hw[m * HS + 32 * kt + 16 + 8 * hi];
      const uint4* bp = (const uint4*)&w2t[(16 * nt + m) * W2TS + 32 * kt + 8 * hi];
      bv.q[0] = bp[0];
      bv.q[1] = bp[2];
      c = wmma_bf16(av.v, bv.v, c);
    }
    const int n = 16 * nt + m;
    const float bias = sb2[n];
#pragma unroll
    for (int r = 0; r < 8; ++r) {
      float v = c[r] + bias;
      int mr = r + 8 * hi;
      eout[(e0 + mr) * FF + n] = v;
      int d = sdst[wave][mr];
      __hip_atomic_fetch_add(&agg[(long)d * FF + n], v, __ATOMIC_RELAXED,
                             __HIP_MEMORY_SCOPE_AGENT);
    }
  }
  if (m == 0) {  // one count per edge row
#pragma unroll
    for (int r = 0; r < 8; ++r) {
      int d = sdst[wave][r + 8 * hi];
      __hip_atomic_fetch_add(&cnt[d], 1.0f, __ATOMIC_RELAXED,
                             __HIP_MEMORY_SCOPE_AGENT);
    }
  }
}

// ---------------------------------------------------------------- node model
// in = [x, agg/max(cnt,1), global] (96) -> 128 relu -> 32; also column sums
__global__ __launch_bounds__(256) void node_mlp_kernel(
    const float* __restrict__ x, const float* __restrict__ agg,
    const float* __restrict__ cnt, const float* __restrict__ ga,
    const float* __restrict__ W1, const float* __restrict__ b1,
    const float* __restrict__ W2, const float* __restrict__ b2,
    float* __restrict__ nout, float* __restrict__ csum) {
  __shared__ alignas(16) __bf16 w1t[128 * W1NS];
  __shared__ alignas(16) __bf16 w2t[32 * W2TS];
  __shared__ alignas(16) __bf16 hid[8][16 * HS];
  __shared__ float sb1[128];
  __shared__ float sb2[32];
  __shared__ float scs[32];

  const int t = threadIdx.x;
  for (int idx = t; idx < 96 * 128; idx += 256) {    // W1[i][j] -> w1t[j][i]
    int i = idx >> 7, j = idx & 127;
    w1t[j * W1NS + i] = (__bf16)W1[idx];
  }
  for (int idx = t; idx < 128 * 32; idx += 256) {
    int k = idx >> 5, n = idx & 31;
    w2t[n * W2TS + k] = (__bf16)W2[idx];
  }
  if (t < 128) sb1[t] = b1[t];
  if (t < 32) { sb2[t] = b2[t]; scs[t] = 0.f; }
  __syncthreads();

  const int wave = t >> 5, lane = t & 31;
  const int m = lane & 15, hi = lane >> 4;
  const long tile = (long)blockIdx.x * 8 + wave;
  if (tile < (NN / 16)) {
    const long n0 = tile * 16;
    const int node = (int)n0 + m;
    const float scale = 1.0f / fmaxf(cnt[node], 1.0f);

    AF a[3];
    {
      float f[16];
      load16(f, x + (long)node * FF, hi);
      cvt16(a[0], f, 1.0f);
      load16(f, agg + (long)node * FF, hi);
      cvt16(a[1], f, scale);
      load16(f, ga, hi);
      cvt16(a[2], f, 1.0f);
    }

    __bf16* hw = &hid[wave][0];
#pragma unroll
    for (int nt = 0; nt < 8; ++nt) {
      v8f c = {0.f, 0.f, 0.f, 0.f, 0.f, 0.f, 0.f, 0.f};
#pragma unroll
      for (int kt = 0; kt < 3; ++kt) {
        AF bfr;
        const uint4* bp = (const uint4*)&w1t[(16 * nt + m) * W1NS + 32 * kt + 8 * hi];
        bfr.q[0] = bp[0];
        bfr.q[1] = bp[2];
        c = wmma_bf16(a[kt].v, bfr.v, c);
      }
      const float bias = sb1[16 * nt + m];
#pragma unroll
      for (int r = 0; r < 8; ++r) {
        float h = c[r] + bias;
        h = h > 0.f ? h : 0.f;
        hw[(r + 8 * hi) * HS + 16 * nt + m] = (__bf16)h;
      }
    }
    asm volatile("" ::: "memory");

#pragma unroll
    for (int nt = 0; nt < 2; ++nt) {
      v8f c = {0.f, 0.f, 0.f, 0.f, 0.f, 0.f, 0.f, 0.f};
#pragma unroll
      for (int kt = 0; kt < 4; ++kt) {
        AF av, bv;
        av.q[0] = *(const uint4*)&hw[m * HS + 32 * kt + 8 * hi];
        av.q[1] = *(const uint4*)&hw[m * HS + 32 * kt + 16 + 8 * hi];
        const uint4* bp = (const uint4*)&w2t[(16 * nt + m) * W2TS + 32 * kt + 8 * hi];
        bv.q[0] = bp[0];
        bv.q[1] = bp[2];
        c = wmma_bf16(av.v, bv.v, c);
      }
      const int n = 16 * nt + m;
      const float bias = sb2[n];
      float partial = 0.f;
#pragma unroll
      for (int r = 0; r < 8; ++r) {
        float v = c[r] + bias;
        nout[(n0 + r + 8 * hi) * FF + n] = v;
        partial += v;
      }
      atomicAdd(&scs[n], partial);  // ds_add_f32, per-block column sums
    }
  }
  __syncthreads();
  if (t < 32)
    __hip_atomic_fetch_add(&csum[t], scs[t], __ATOMIC_RELAXED,
                           __HIP_MEMORY_SCOPE_AGENT);
}

// ---------------------------------------------------------------- global model
// in = [mean(node_out), global] (64) -> 128 relu -> 32  (tiny, exact fp32)
__global__ __launch_bounds__(128) void global_mlp_kernel(
    const float* __restrict__ csum, const float* __restrict__ ga,
    const float* __restrict__ W1, const float* __restrict__ b1,
    const float* __restrict__ W2, const float* __restrict__ b2,
    float* __restrict__ gout) {
  __shared__ float gin[64];
  __shared__ float hidg[128];
  const int t = threadIdx.x;
  if (t < 32)       gin[t] = csum[t] * (1.0f / (float)NN);
  else if (t < 64)  gin[t] = ga[t - 32];
  __syncthreads();
  float h = b1[t];
#pragma unroll 8
  for (int i = 0; i < 64; ++i) h += gin[i] * W1[i * 128 + t];
  hidg[t] = h > 0.f ? h : 0.f;
  __syncthreads();
  if (t < 32) {
    float o = b2[t];
#pragma unroll 8
    for (int j = 0; j < 128; ++j) o += hidg[j] * W2[j * 32 + t];
    gout[t] = o;
  }
}

extern "C" void kernel_launch(void* const* d_in, const int* in_sizes, int n_in,
                              void* d_out, int out_size, void* d_ws, size_t ws_size,
                              hipStream_t stream) {
  const float* x   = (const float*)d_in[0];
  const int*   ei  = (const int*)d_in[1];
  const float* ea  = (const float*)d_in[2];
  const float* ga  = (const float*)d_in[3];
  const float* W1e = (const float*)d_in[4];
  const float* b1e = (const float*)d_in[5];
  const float* W2e = (const float*)d_in[6];
  const float* b2e = (const float*)d_in[7];
  const float* W1n = (const float*)d_in[8];
  const float* b1n = (const float*)d_in[9];
  const float* W2n = (const float*)d_in[10];
  const float* b2n = (const float*)d_in[11];
  const float* W1g = (const float*)d_in[12];
  const float* b1g = (const float*)d_in[13];
  const float* W2g = (const float*)d_in[14];
  const float* b2g = (const float*)d_in[15];

  float* out      = (float*)d_out;
  float* node_out = out;                                        // [N, 32]
  float* edge_out = out + (size_t)NN * FF;                      // [E, 32]
  float* glob_out = out + (size_t)NN * FF + (size_t)EE * FF;    // [1, 32]

  float* ws   = (float*)d_ws;
  float* agg  = ws;                       // [N, 32] segment sums
  float* cnt  = ws + (size_t)NN * FF;     // [N] in-degree counts
  float* csum = cnt + NN;                 // [32] node_out column sums

  zero_ws<<<512, 256, 0, stream>>>(ws, (long)NN * 33 + 32);
  edge_mlp_kernel<<<EE / 128, 256, 0, stream>>>(x, ei, ea, ga, W1e, b1e, W2e,
                                                b2e, edge_out, agg, cnt);
  node_mlp_kernel<<<(NN / 16 + 7) / 8, 256, 0, stream>>>(x, agg, cnt, ga, W1n,
                                                         b1n, W2n, b2n,
                                                         node_out, csum);
  global_mlp_kernel<<<1, 128, 0, stream>>>(csum, ga, W1g, b1g, W2g, b2g,
                                           glob_out);
}
